// MultiHeadAttention_55533927137626
// MI455X (gfx1250) — matrix operand order repaired and checked by reading, not compiled, not run on hardware
//
#include <hip/hip_runtime.h>
#include <hip/hip_bf16.h>

// MultiHeadAttention forward for MI455X (gfx1250), bf16 WMMA pipeline with
// async global->LDS staging (double buffered). B=4, S=2048, H=1024, NH=16, HD=64.

#define B_   4
#define S_   2048
#define H_   1024
#define NH_  16
#define HD_  64
#define BS_  (B_ * S_)

typedef __bf16 bf16;
typedef __attribute__((ext_vector_type(16))) __bf16 v16bf;
typedef __attribute__((ext_vector_type(8)))  __bf16 bf16x8;
typedef __attribute__((ext_vector_type(4)))  __bf16 bf16x4;
typedef __attribute__((ext_vector_type(8)))  float  v8f;

// ---- CDNA5 async global->LDS copy (ASYNCcnt-tracked), inline asm form -------
// dsaddr = low 32 bits of generic shared pointer (flat->LDS aperture truncation)
#define ASYNC_LD_B128(ldsoff, gaddr)                                         \
    asm volatile("global_load_async_to_lds_b128 %0, %1, off"                 \
                 :: "v"((unsigned)(ldsoff)), "v"(gaddr) : "memory")
#define WAIT_ASYNC(n) asm volatile("s_wait_asynccnt " #n ::: "memory")

static __device__ inline unsigned lds_off(const void* p) {
    return (unsigned)(size_t)p;
}

static __device__ inline v8f zero_v8f() {
    v8f z;
    for (int i = 0; i < 8; ++i) z[i] = 0.0f;
    return z;
}

// A-fragment (16x32 bf16): per-lane 16 contiguous K elements
static __device__ inline v16bf ld_afrag(const bf16* p) {
    bf16x8 lo = *(const bf16x8*)(p);
    bf16x8 hi = *(const bf16x8*)(p + 8);
    return __builtin_shufflevector(lo, hi, 0,1,2,3,4,5,6,7,8,9,10,11,12,13,14,15);
}
// B-fragment (32x16 bf16): per-lane 16 contiguous K elements
static __device__ inline v16bf ld_bfrag(const bf16* p) {
    bf16x8 lo = *(const bf16x8*)(p);
    bf16x8 hi = *(const bf16x8*)(p + 8);
    return __builtin_shufflevector(lo, hi, 0,1,2,3,4,5,6,7,8,9,10,11,12,13,14,15);
}
static __device__ inline v8f wmma_bf16(v16bf a, v16bf b, v8f c) {
    return __builtin_amdgcn_wmma_f32_16x16x32_bf16(false, a, false, b, (short)0, c,
                                                   false, false);
}

static __device__ inline float half_reduce_max(float v) {
    v = fmaxf(v, __shfl_xor(v, 1, 32));
    v = fmaxf(v, __shfl_xor(v, 2, 32));
    v = fmaxf(v, __shfl_xor(v, 4, 32));
    v = fmaxf(v, __shfl_xor(v, 8, 32));
    return v;
}
static __device__ inline float half_reduce_sum(float v) {
    v += __shfl_xor(v, 1, 32);
    v += __shfl_xor(v, 2, 32);
    v += __shfl_xor(v, 4, 32);
    v += __shfl_xor(v, 8, 32);
    return v;
}

// ---------------------------------------------------------------------------
// f32 -> bf16 conversion (vectorized x4)
// ---------------------------------------------------------------------------
__global__ __launch_bounds__(256) void cvt_f32_bf16_kernel(const float* __restrict__ in,
                                                           bf16* __restrict__ out,
                                                           int n4) {
    int i = blockIdx.x * blockDim.x + threadIdx.x;
    if (i < n4) {
        float4 v = ((const float4*)in)[i];
        bf16x4 o;
        o[0] = (bf16)v.x; o[1] = (bf16)v.y; o[2] = (bf16)v.z; o[3] = (bf16)v.w;
        ((bf16x4*)out)[i] = o;
    }
}

// ---------------------------------------------------------------------------
// f32 W[k][n] -> bf16 WT[n][k]  (H_ x H_), tiled transpose through LDS.
// Pre-transposing once makes every GEMM LDS fill a contiguous async copy.
// ---------------------------------------------------------------------------
__global__ __launch_bounds__(256) void cvt_transpose_kernel(const float* __restrict__ in,
                                                            bf16* __restrict__ out) {
    __shared__ __align__(16) bf16 Tt[64 * 72];
    const int t  = threadIdx.x;
    const int n0 = blockIdx.x * 64;
    const int k0 = blockIdx.y * 64;
#pragma unroll
    for (int i = 0; i < 4; ++i) {
        int id = t + i * 256;            // 0..1023
        int kk = id >> 4;                // 0..63
        int nc = id & 15;                // float4 chunk
        float4 v = *(const float4*)(in + (size_t)(k0 + kk) * H_ + n0 + nc * 4);
        bf16x4 o;
        o[0] = (bf16)v.x; o[1] = (bf16)v.y; o[2] = (bf16)v.z; o[3] = (bf16)v.w;
        *(bf16x4*)&Tt[kk * 72 + nc * 4] = o;      // Tt[k][n]
    }
    __syncthreads();
#pragma unroll
    for (int i = 0; i < 2; ++i) {
        int id = t + i * 256;            // 0..511
        int nn = id >> 3;                // 0..63
        int c  = id & 7;                 // 8-elem k chunk
        bf16x8 o;
#pragma unroll
        for (int q = 0; q < 8; ++q) o[q] = Tt[(c * 8 + q) * 72 + nn];
        *(bf16x8*)(out + (size_t)(n0 + nn) * H_ + k0 + c * 8) = o;
    }
}

// ---------------------------------------------------------------------------
// GEMM: Y[M=8192, N=1024] = X[M,K=1024] * W[K,N] + bias,  W given transposed
// (WT[n][k]).  Async double-buffered LDS staging, k-step 32.
// mode 0: bf16 out, (B,NH,S,HD) layout     (Q, K projections)
// mode 2: bf16 out, (B,NH,HD,S) layout     (V projection, pre-transposed)
// mode 1: f32 out, row-major d_out + bias  (final projection)
// ---------------------------------------------------------------------------
#define GM_TILE 128
#define GN_TILE 64
#define GK_TILE 32
#define XS_STR  40
#define WT_STR  40

__global__ __launch_bounds__(256) void gemm_bf16_kernel(const bf16* __restrict__ X,
                                                        const bf16* __restrict__ WT,
                                                        const float* __restrict__ bias,
                                                        bf16* __restrict__ outq,
                                                        float* __restrict__ outf,
                                                        int mode) {
    __shared__ __align__(16) bf16 Xs[2][GM_TILE * XS_STR];
    __shared__ __align__(16) bf16 Ws[2][GN_TILE * WT_STR];

    const int t    = threadIdx.x;
    const int w    = t >> 5;
    const int lane = t & 31;
    const int half = lane >> 4;
    const int col0 = lane & 15;
    const int m0   = blockIdx.y * GM_TILE;
    const int n0   = blockIdx.x * GN_TILE;

    // staging coordinates: X tile 128x32 (512 x 16B chunks, 2/thread),
    //                      W tile  64x32 (256 x 16B chunks, 1/thread)
    const int xr = t >> 2, xc = t & 3;
    const bf16* xsrc0 = X + (size_t)(m0 + xr) * H_ + xc * 8;
    const bf16* xsrc1 = X + (size_t)(m0 + xr + 64) * H_ + xc * 8;
    const bf16* wsrc  = WT + (size_t)(n0 + xr) * H_ + xc * 8;   // xr:0..63 row, xc chunk

    unsigned xdst0[2], xdst1[2], wdst[2];
#pragma unroll
    for (int s = 0; s < 2; ++s) {
        xdst0[s] = lds_off(&Xs[s][xr * XS_STR + xc * 8]);
        xdst1[s] = lds_off(&Xs[s][(xr + 64) * XS_STR + xc * 8]);
        wdst[s]  = lds_off(&Ws[s][xr * WT_STR + xc * 8]);
    }

    auto stage = [&](int s, int k0) {
        ASYNC_LD_B128(xdst0[s], xsrc0 + k0);
        ASYNC_LD_B128(xdst1[s], xsrc1 + k0);
        ASYNC_LD_B128(wdst[s],  wsrc  + k0);
    };

    v8f acc[4];
#pragma unroll
    for (int i = 0; i < 4; ++i) acc[i] = zero_v8f();

    stage(0, 0);
    for (int k0 = 0, it = 0; k0 < H_; k0 += GK_TILE, ++it) {
        const int cur = it & 1;
        if (k0 + GK_TILE < H_) {
            stage(cur ^ 1, k0 + GK_TILE);
            WAIT_ASYNC(3);
        } else {
            WAIT_ASYNC(0);
        }
        __syncthreads();

        v16bf a = ld_afrag(&Xs[cur][(w * 16 + col0) * XS_STR + half * 16]);
#pragma unroll
        for (int tt = 0; tt < 4; ++tt) {
            v16bf b = ld_bfrag(&Ws[cur][(tt * 16 + col0) * WT_STR + half * 16]);
            acc[tt] = wmma_bf16(a, b, acc[tt]);
        }
        __syncthreads();   // protect buffer reuse before next async issue
    }

#pragma unroll
    for (int tt = 0; tt < 4; ++tt) {
        int n = n0 + tt * 16 + col0;
        float bv = bias[n];
#pragma unroll
        for (int r = 0; r < 8; ++r) {
            int m = m0 + w * 16 + half * 8 + r;
            float v = acc[tt][r] + bv;
            int bb = m / S_, ss = m % S_;
            int hh = n / HD_, dd = n % HD_;
            if (mode == 0) {
                outq[(((size_t)bb * NH_ + hh) * S_ + ss) * HD_ + dd] = (bf16)v;
            } else if (mode == 2) {
                outq[(((size_t)bb * NH_ + hh) * HD_ + dd) * S_ + ss] = (bf16)v;
            } else {
                outf[(size_t)m * H_ + n] = v;
            }
        }
    }
}

// ---------------------------------------------------------------------------
// Attention: block = (b, h, 64-query tile), 128 threads (4 waves).
// Two-sweep online softmax, async double-buffered K / V^T tiles.
// V comes in pre-transposed (B,NH,HD,S) so all staging is contiguous.
// ---------------------------------------------------------------------------
#define QS_STR 72
#define PS_STR 72
#define SCALE_ 0.125f   // 1/sqrt(64)

static __device__ inline void score_tile(v8f sc[4], const bf16* Qs, const bf16* Ks,
                                         const int* mrow, int j, int w, int half,
                                         int col0) {
    v16bf a0 = ld_afrag(&Qs[(w * 16 + col0) * QS_STR + 0 + half * 16]);
    v16bf a1 = ld_afrag(&Qs[(w * 16 + col0) * QS_STR + 32 + half * 16]);
#pragma unroll
    for (int tt = 0; tt < 4; ++tt) {
        v16bf b0 = ld_bfrag(&Ks[(tt * 16 + col0) * QS_STR + 0 + half * 16]);
        v16bf b1 = ld_bfrag(&Ks[(tt * 16 + col0) * QS_STR + 32 + half * 16]);
        v8f c = zero_v8f();
        c = wmma_bf16(a0, b0, c);
        c = wmma_bf16(a1, b1, c);
        int key = j * 64 + tt * 16 + col0;
        int mv = mrow[key];
#pragma unroll
        for (int r = 0; r < 8; ++r) {
            float s = c[r] * SCALE_;
            sc[tt][r] = (mv == 0) ? -1.0e9f : s;
        }
    }
}

__global__ __launch_bounds__(128) void attn_kernel(const bf16* __restrict__ Q,
                                                   const bf16* __restrict__ K,
                                                   const bf16* __restrict__ VT,
                                                   const int* __restrict__ mask,
                                                   float* __restrict__ attn,
                                                   bf16* __restrict__ ctx) {
    __shared__ __align__(16) bf16 Qs[64 * QS_STR];
    __shared__ __align__(16) bf16 Ks[2][64 * QS_STR];
    __shared__ __align__(16) bf16 Vs[2][64 * QS_STR];   // [d][key]
    __shared__ __align__(16) bf16 Ps[4][16 * PS_STR];

    const int t    = threadIdx.x;
    const int w    = t >> 5;
    const int lane = t & 31;
    const int half = lane >> 4;
    const int col0 = lane & 15;
    const int b    = blockIdx.z;
    const int h    = blockIdx.y;
    const int q0   = blockIdx.x * 64;
    const size_t bh = (size_t)b * NH_ + h;

    const bf16* Qbase = Q + (bh * S_ + q0) * HD_;
    const bf16* Kbase = K + bh * S_ * HD_;
    const bf16* Vbase = VT + bh * HD_ * S_;
    const int*  mrow  = mask + (size_t)b * S_;

    // staging coordinates: 64 rows x 128B => 512 chunks, 4/thread
    int srow[4], scol[4];
    unsigned kdst[2][4], vdst[2][4];
#pragma unroll
    for (int i = 0; i < 4; ++i) {
        int id = t + i * 128;
        srow[i] = id >> 3;
        scol[i] = id & 7;
#pragma unroll
        for (int s = 0; s < 2; ++s) {
            kdst[s][i] = lds_off(&Ks[s][srow[i] * QS_STR + scol[i] * 8]);
            vdst[s][i] = lds_off(&Vs[s][srow[i] * QS_STR + scol[i] * 8]);
        }
    }
    auto stageK = [&](int s, int j) {
#pragma unroll
        for (int i = 0; i < 4; ++i)
            ASYNC_LD_B128(kdst[s][i],
                          Kbase + ((size_t)(j * 64 + srow[i])) * HD_ + scol[i] * 8);
    };
    auto stageV = [&](int s, int j) {
#pragma unroll
        for (int i = 0; i < 4; ++i)
            ASYNC_LD_B128(vdst[s][i],
                          Vbase + (size_t)srow[i] * S_ + j * 64 + scol[i] * 8);
    };

    // stage Q tile once (regular vector copy; covered by first barrier)
#pragma unroll
    for (int i = 0; i < 4; ++i) {
        *(bf16x8*)&Qs[srow[i] * QS_STR + scol[i] * 8] =
            *(const bf16x8*)(Qbase + srow[i] * HD_ + scol[i] * 8);
    }

    float rm[8], rl[8];
#pragma unroll
    for (int r = 0; r < 8; ++r) { rm[r] = -3.0e38f; rl[r] = 0.0f; }

    // ---------------- sweep 1: softmax statistics ----------------
    stageK(0, 0);
    for (int j = 0; j < S_ / 64; ++j) {
        const int cur = j & 1;
        if (j + 1 < S_ / 64) { stageK(cur ^ 1, j + 1); WAIT_ASYNC(4); }
        else                 { WAIT_ASYNC(0); }
        __syncthreads();

        v8f sc[4];
        score_tile(sc, Qs, Ks[cur], mrow, j, w, half, col0);

#pragma unroll
        for (int r = 0; r < 8; ++r) {
            float tm = sc[0][r];
            tm = fmaxf(tm, sc[1][r]);
            tm = fmaxf(tm, sc[2][r]);
            tm = fmaxf(tm, sc[3][r]);
            tm = half_reduce_max(tm);
            float mn = fmaxf(rm[r], tm);
            float ps = __expf(sc[0][r] - mn) + __expf(sc[1][r] - mn) +
                       __expf(sc[2][r] - mn) + __expf(sc[3][r] - mn);
            ps = half_reduce_sum(ps);
            rl[r] = rl[r] * __expf(rm[r] - mn) + ps;
            rm[r] = mn;
        }
        __syncthreads();
    }

    float rli[8];
#pragma unroll
    for (int r = 0; r < 8; ++r) rli[r] = (rl[r] > 0.0f) ? (1.0f / rl[r]) : 0.0f;

    // ---------------- sweep 2: probs out + ctx = p @ V ----------------
    v8f o[4];
#pragma unroll
    for (int i = 0; i < 4; ++i) o[i] = zero_v8f();

    stageK(0, 0);
    stageV(0, 0);
    for (int j = 0; j < S_ / 64; ++j) {
        const int cur = j & 1;
        if (j + 1 < S_ / 64) { stageK(cur ^ 1, j + 1); stageV(cur ^ 1, j + 1); WAIT_ASYNC(8); }
        else                 { WAIT_ASYNC(0); }
        __syncthreads();

        v8f sc[4];
        score_tile(sc, Qs, Ks[cur], mrow, j, w, half, col0);

#pragma unroll
        for (int tt = 0; tt < 4; ++tt) {
            int key = j * 64 + tt * 16 + col0;
#pragma unroll
            for (int r = 0; r < 8; ++r) {
                float p = __expf(sc[tt][r] - rm[r]) * rli[r];
                int q = q0 + w * 16 + half * 8 + r;
                attn[(bh * S_ + q) * S_ + key] = p;
                Ps[w][(half * 8 + r) * PS_STR + tt * 16 + col0] = (bf16)p;
            }
        }
        __syncthreads();

#pragma unroll
        for (int ks = 0; ks < 2; ++ks) {
            v16bf a = ld_afrag(&Ps[w][col0 * PS_STR + ks * 32 + half * 16]);
#pragma unroll
            for (int tt = 0; tt < 4; ++tt) {
                v16bf bb = ld_bfrag(&Vs[cur][(tt * 16 + col0) * QS_STR + ks * 32 + half * 16]);
                o[tt] = wmma_bf16(a, bb, o[tt]);
            }
        }
        __syncthreads();   // protect Ks/Vs buffer reuse before next async issue
    }

    // ctx in (b, s, h*HD+d) layout for the output projection
#pragma unroll
    for (int tt = 0; tt < 4; ++tt) {
#pragma unroll
        for (int r = 0; r < 8; ++r) {
            int q = q0 + w * 16 + half * 8 + r;
            ctx[((size_t)b * S_ + q) * H_ + h * HD_ + tt * 16 + col0] = (bf16)o[tt][r];
        }
    }
}

// ---------------------------------------------------------------------------
// Host-side launcher
// ---------------------------------------------------------------------------
extern "C" void kernel_launch(void* const* d_in, const int* in_sizes, int n_in,
                              void* d_out, int out_size, void* d_ws, size_t ws_size,
                              hipStream_t stream) {
    (void)in_sizes; (void)n_in; (void)out_size; (void)ws_size;

    const float* hs    = (const float*)d_in[0];
    const int*   amask = (const int*)d_in[1];
    const float* Wq    = (const float*)d_in[2];
    const float* bq    = (const float*)d_in[3];
    const float* Wk    = (const float*)d_in[4];
    const float* bk    = (const float*)d_in[5];
    const float* Wv    = (const float*)d_in[6];
    const float* bv    = (const float*)d_in[7];
    const float* Wo    = (const float*)d_in[8];
    const float* bo    = (const float*)d_in[9];

    float* out  = (float*)d_out;                       // B*S*H
    float* attn = out + (size_t)BS_ * H_;              // B*NH*S*S

    char* ws = (char*)d_ws;
    bf16* Xb  = (bf16*)ws; ws += (size_t)BS_ * H_ * sizeof(bf16);
    bf16* Wqb = (bf16*)ws; ws += (size_t)H_ * H_ * sizeof(bf16);   // transposed
    bf16* Wkb = (bf16*)ws; ws += (size_t)H_ * H_ * sizeof(bf16);   // transposed
    bf16* Wvb = (bf16*)ws; ws += (size_t)H_ * H_ * sizeof(bf16);   // transposed
    bf16* Wob = (bf16*)ws; ws += (size_t)H_ * H_ * sizeof(bf16);   // transposed
    bf16* Qb  = (bf16*)ws; ws += (size_t)BS_ * H_ * sizeof(bf16);
    bf16* Kb  = (bf16*)ws; ws += (size_t)BS_ * H_ * sizeof(bf16);
    bf16* VTb = (bf16*)ws; ws += (size_t)BS_ * H_ * sizeof(bf16);
    bf16* Cb  = (bf16*)ws; ws += (size_t)BS_ * H_ * sizeof(bf16);

    const int thr = 256;
    {
        int n4 = BS_ * H_ / 4;
        cvt_f32_bf16_kernel<<<(n4 + thr - 1) / thr, thr, 0, stream>>>(hs, Xb, n4);
    }
    {
        dim3 tg(H_ / 64, H_ / 64);
        cvt_transpose_kernel<<<tg, thr, 0, stream>>>(Wq, Wqb);
        cvt_transpose_kernel<<<tg, thr, 0, stream>>>(Wk, Wkb);
        cvt_transpose_kernel<<<tg, thr, 0, stream>>>(Wv, Wvb);
        cvt_transpose_kernel<<<tg, thr, 0, stream>>>(Wo, Wob);
    }

    dim3 gg(H_ / GN_TILE, BS_ / GM_TILE);   // 16 x 64
    gemm_bf16_kernel<<<gg, 256, 0, stream>>>(Xb, Wqb, bq, Qb, nullptr, 0);
    gemm_bf16_kernel<<<gg, 256, 0, stream>>>(Xb, Wkb, bk, Kb, nullptr, 0);
    gemm_bf16_kernel<<<gg, 256, 0, stream>>>(Xb, Wvb, bv, VTb, nullptr, 2);

    dim3 ag(S_ / 64, NH_, B_);
    attn_kernel<<<ag, 128, 0, stream>>>(Qb, Kb, VTb, amask, attn, Cb);

    gemm_bf16_kernel<<<gg, 256, 0, stream>>>(Cb, Wob, bo, nullptr, out, 1);
}
